// GenericGraphMoE_67783173865960
// MI455X (gfx1250) — matrix operand order repaired and checked by reading, not compiled, not run on hardware
//
#include <hip/hip_runtime.h>
#include <hip/hip_bf16.h>

#define N_NODES 100000
#define N_EDGES 1600000
#define HID 128
#define NGRAPH 64
#define PRIOR_W 0.35f

typedef __attribute__((ext_vector_type(16))) _Float16 v16h;
typedef __attribute__((ext_vector_type(8)))  float    v8f;

// ---------------- graph stats ----------------
__global__ __launch_bounds__(256) void count_nodes_kernel(const int* __restrict__ batch,
                                                          float* __restrict__ ncnt) {
  int i = blockIdx.x * 256 + threadIdx.x;
  if (i < N_NODES) atomicAdd(&ncnt[batch[i]], 1.0f);
}

__global__ __launch_bounds__(256) void count_edges_kernel(const int* __restrict__ src,
                                                          const int* __restrict__ batch,
                                                          float* __restrict__ ecnt) {
  int e = blockIdx.x * 256 + threadIdx.x;
  if (e < N_EDGES) atomicAdd(&ecnt[batch[src[e]]], 1.0f);
}

__global__ __launch_bounds__(64) void graph_stats_kernel(const float* __restrict__ ncnt,
                                                         const float* __restrict__ ecnt,
                                                         float* __restrict__ gfn,   // [G,3]
                                                         float* __restrict__ lnn) { // [G]
  __shared__ float nn[NGRAPH], ee[NGRAPH], dd[NGRAPH], ln[NGRAPH];
  int g = threadIdx.x;
  float n = fmaxf(ncnt[g], 1.0f);
  float e = ecnt[g];
  float dens = e / fmaxf(n * (n - 1.0f), 1.0f);
  nn[g] = n; ee[g] = e; dd[g] = dens; ln[g] = logf(n);
  __syncthreads();
  if (g == 0) {
    float mn = ln[0], mx = ln[0];
    for (int i = 1; i < NGRAPH; ++i) { mn = fminf(mn, ln[i]); mx = fmaxf(mx, ln[i]); }
    float mean0 = 0, mean1 = 0, mean2 = 0;
    for (int i = 0; i < NGRAPH; ++i) { mean0 += nn[i]; mean1 += ee[i]; mean2 += dd[i]; }
    mean0 /= (float)NGRAPH; mean1 /= (float)NGRAPH; mean2 /= (float)NGRAPH;
    float v0 = 0, v1 = 0, v2 = 0;
    for (int i = 0; i < NGRAPH; ++i) {
      float d0 = nn[i] - mean0, d1 = ee[i] - mean1, d2 = dd[i] - mean2;
      v0 += d0 * d0; v1 += d1 * d1; v2 += d2 * d2;
    }
    float s0 = sqrtf(v0 / (float)NGRAPH) + 1e-6f;
    float s1 = sqrtf(v1 / (float)NGRAPH) + 1e-6f;
    float s2 = sqrtf(v2 / (float)NGRAPH) + 1e-6f;
    float inv = 1.0f / (mx - mn + 1e-6f);
    for (int i = 0; i < NGRAPH; ++i) {
      gfn[i * 3 + 0] = (nn[i] - mean0) / s0;
      gfn[i * 3 + 1] = (ee[i] - mean1) / s1;
      gfn[i * 3 + 2] = (dd[i] - mean2) / s2;
      lnn[i] = (ln[i] - mn) * inv;
    }
  }
}

// ---------------- encoder: h = relu(x6 @ W1 + b1) @ W2 + b2 ----------------
__global__ __launch_bounds__(128) void encoder_kernel(
    const float* __restrict__ x, const float* __restrict__ W1, const float* __restrict__ b1,
    const float* __restrict__ W2, const float* __restrict__ b2, float* __restrict__ h) {
  __shared__ float xl[6];
  __shared__ float hid[HID];
  int n = blockIdx.x, t = threadIdx.x;
  if (t < 6) xl[t] = x[(size_t)n * 16 + 4 + t];
  __syncthreads();
  float s = b1[t];
#pragma unroll
  for (int k = 0; k < 6; ++k) s += xl[k] * W1[k * HID + t];
  hid[t] = fmaxf(s, 0.0f);
  __syncthreads();
  float o = b2[t];
  for (int j = 0; j < HID; ++j) o += hid[j] * W2[j * HID + t];
  h[(size_t)n * HID + t] = o;
}

// ---------------- router + top-2 gate ----------------
__global__ __launch_bounds__(128) void router_kernel(
    const float* __restrict__ h, const int* __restrict__ batch,
    const float* __restrict__ gfn, const float* __restrict__ lnn,
    const float* __restrict__ W1, const float* __restrict__ b1,
    const float* __restrict__ W2, const float* __restrict__ b2,
    const float* __restrict__ centers, float* __restrict__ wout) {
  __shared__ float cv[131];
  __shared__ float hid[HID];
  __shared__ float lg[4];
  int n = blockIdx.x, t = threadIdx.x;
  int g = batch[n];
  cv[t] = h[(size_t)n * HID + t];
  if (t < 3) cv[HID + t] = gfn[g * 3 + t];
  __syncthreads();
  float s = b1[t];
  for (int k = 0; k < 131; ++k) s += cv[k] * W1[k * HID + t];
  hid[t] = fmaxf(s, 0.0f);
  __syncthreads();
  if (t < 4) {
    float s2 = b2[t];
    for (int k = 0; k < HID; ++k) s2 += hid[k] * W2[k * 4 + t];
    float d = lnn[g] - centers[t];
    lg[t] = (1.0f - PRIOR_W) * s2 + PRIOR_W * (-(d * d));
  }
  __syncthreads();
  if (t == 0) {
    float m = fmaxf(fmaxf(lg[0], lg[1]), fmaxf(lg[2], lg[3]));
    float p[4]; float sum = 0.0f;
    for (int i = 0; i < 4; ++i) { p[i] = expf(lg[i] - m); sum += p[i]; }
    for (int i = 0; i < 4; ++i) p[i] /= sum;
    int i0 = 0;
    for (int i = 1; i < 4; ++i) if (p[i] > p[i0]) i0 = i;
    int i1 = -1;
    for (int i = 0; i < 4; ++i) { if (i == i0) continue; if (i1 < 0 || p[i] > p[i1]) i1 = i; }
    float inv = 1.0f / (p[i0] + p[i1] + 1e-8f);
    float w4[4] = {0.f, 0.f, 0.f, 0.f};
    w4[i0] = p[i0] * inv; w4[i1] = p[i1] * inv;
    float* o = wout + (size_t)n * 4;
    o[0] = w4[0]; o[1] = w4[1]; o[2] = w4[2]; o[3] = w4[3];
  }
}

// ---------------- weight pack: f32 [128(K),128(N)] -> WMMA B-fragments f16 ----------------
// packed[m][w(8)][c(4)][lane(32)][j(16)] ; k = c*32 + (lane>=16?8:0) + (j&7) + (j>=8?16:0) ; n = w*16 + (lane&15)
__global__ __launch_bounds__(256) void pack_weights_kernel(
    const float* __restrict__ start_rel, const float* __restrict__ start_root,
    const float* __restrict__ mid_rel, const float* __restrict__ mid_root,
    _Float16* __restrict__ packed) {
  int t = blockIdx.x * 256 + threadIdx.x;  // 16*16384 threads
  int m = t >> 14;
  int idx = t & 16383;
  int e = m >> 2, which = m & 3;
  const float* srcm;
  switch (which) {
    case 0:  srcm = start_rel  + e * 16384; break;
    case 1:  srcm = start_root + e * 16384; break;
    case 2:  srcm = mid_rel    + e * 16384; break;
    default: srcm = mid_root   + e * 16384; break;
  }
  int j = idx & 15;
  int l = (idx >> 4) & 31;
  int c = (idx >> 9) & 3;
  int w = (idx >> 11) & 7;
  int n = w * 16 + (l & 15);
  int k = c * 32 + ((l >> 4) * 8) + (j & 7) + ((j >> 3) * 16);
  packed[(size_t)m * 16384 + idx] = (_Float16)srcm[k * HID + n];
}

// ---------------- edge scatter: agg[dst] += feat[src] (f32 atomics, L2-resident) ----------------
__global__ __launch_bounds__(256) void agg_scatter_kernel(
    const float* __restrict__ feat, float* __restrict__ agg,
    const int* __restrict__ src, const int* __restrict__ dst) {
  unsigned t = blockIdx.x * 256u + threadIdx.x;
  int edge = (int)(t >> 5);
  int cb = (int)(t & 31u) * 4;
  if (edge >= N_EDGES) return;
  int s = src[edge], d = dst[edge];
  const float4 v = *(const float4*)(feat + (size_t)s * HID + cb);
  float* a = agg + (size_t)d * HID + cb;
  atomicAdd(a + 0, v.x);
  atomicAdd(a + 1, v.y);
  atomicAdd(a + 2, v.z);
  atomicAdd(a + 3, v.w);
}

// ---------------- dual GEMM with WMMA: D = act(A1@B1 + A2@B2 + bias) ----------------
// A lane layout (ISA 7.12.2, 16-bit A 16x32): lane l row=l&15,
//   elems 0..7 = A[r][base..base+8), elems 8..15 = A[r][base+16..base+24), base=c*32+(l>=16?8:0)
__device__ inline v16h cvt_a_frag(const float* __restrict__ rowp, int base) {
  const float4* p0 = (const float4*)(rowp + base);
  const float4* p1 = (const float4*)(rowp + base + 16);
  float4 x0 = p0[0], x1 = p0[1], y0 = p1[0], y1 = p1[1];
  v16h a;
  a[0] = (_Float16)x0.x; a[1] = (_Float16)x0.y; a[2]  = (_Float16)x0.z; a[3]  = (_Float16)x0.w;
  a[4] = (_Float16)x1.x; a[5] = (_Float16)x1.y; a[6]  = (_Float16)x1.z; a[7]  = (_Float16)x1.w;
  a[8] = (_Float16)y0.x; a[9] = (_Float16)y0.y; a[10] = (_Float16)y0.z; a[11] = (_Float16)y0.w;
  a[12] = (_Float16)y1.x; a[13] = (_Float16)y1.y; a[14] = (_Float16)y1.z; a[15] = (_Float16)y1.w;
  return a;
}

// Each wave owns one 16-row tile and ALL 8 column tiles (8 C fragments):
// A fragments are loaded/converted once and reused across 8 WMMAs each.
__global__ __launch_bounds__(256) void gemm_dual_kernel(
    const float* __restrict__ A1, const _Float16* __restrict__ B1,
    const float* __restrict__ A2, const _Float16* __restrict__ B2,
    const float* __restrict__ bias, float* __restrict__ D, int do_relu) {
  const int wave = threadIdx.x >> 5;
  const int lane = threadIdx.x & 31;
  const int rowTile = blockIdx.x * 8 + wave;
  if (rowTile >= N_NODES / 16) return;
  const int r = rowTile * 16 + (lane & 15);
  const float* row1 = A1 + (size_t)r * HID;
  const float* row2 = A2 + (size_t)r * HID;
  const int hi8 = (lane >> 4) * 8;
  v8f acc[8];
#pragma unroll
  for (int i = 0; i < 8; ++i) acc[i] = (v8f){};
#pragma unroll
  for (int c = 0; c < 4; ++c) {
    const int base = c * 32 + hi8;
    v16h af1 = cvt_a_frag(row1, base);
    v16h af2 = cvt_a_frag(row2, base);
#pragma unroll
    for (int w8 = 0; w8 < 8; ++w8) {
      v16h bf1 = *(const v16h*)(B1 + ((size_t)((w8 * 4 + c) * 32 + lane)) * 16);
      acc[w8] = __builtin_amdgcn_wmma_f32_16x16x32_f16(false, af1, false, bf1, (short)0,
                                                       acc[w8], false, false);
      v16h bf2 = *(const v16h*)(B2 + ((size_t)((w8 * 4 + c) * 32 + lane)) * 16);
      acc[w8] = __builtin_amdgcn_wmma_f32_16x16x32_f16(false, af2, false, bf2, (short)0,
                                                       acc[w8], false, false);
    }
  }
  const int rbase = rowTile * 16 + hi8;  // C/D: lanes>=16 hold rows 8..15 of tile
#pragma unroll
  for (int w8 = 0; w8 < 8; ++w8) {
    const int col = w8 * 16 + (lane & 15);
    const float bv = bias[col];
#pragma unroll
    for (int j = 0; j < 8; ++j) {
      float v = acc[w8][j] + bv;
      if (do_relu) v = fmaxf(v, 0.0f);
      D[(size_t)(rbase + j) * HID + col] = v;
    }
  }
}

// ---------------- end layer (128->4) fused with gate blend ----------------
__global__ __launch_bounds__(256) void end_layer_kernel(
    const float* __restrict__ agg, const float* __restrict__ z,
    const float* __restrict__ relW, const float* __restrict__ relB,
    const float* __restrict__ rootW, const float* __restrict__ wgt,
    int expert, float* __restrict__ pred) {
  int n = blockIdx.x * 256 + threadIdx.x;
  if (n >= N_NODES) return;
  float w = wgt[(size_t)n * 4 + expert];
  if (w == 0.0f) return;  // top-2 gate: contribution is exactly zero
  float o0 = relB[0], o1 = relB[1], o2 = relB[2], o3 = relB[3];
  const float* ar = agg + (size_t)n * HID;
  const float* zr = z + (size_t)n * HID;
  for (int k = 0; k < HID; ++k) {
    float av = ar[k], zv = zr[k];
    const float4 rw = *(const float4*)(relW + k * 4);
    const float4 ow = *(const float4*)(rootW + k * 4);
    o0 += av * rw.x + zv * ow.x;
    o1 += av * rw.y + zv * ow.y;
    o2 += av * rw.z + zv * ow.z;
    o3 += av * rw.w + zv * ow.w;
  }
  float* p = pred + (size_t)n * 4;
  p[0] += w * o0; p[1] += w * o1; p[2] += w * o2; p[3] += w * o3;
}

// ---------------- host orchestration ----------------
extern "C" void kernel_launch(void* const* d_in, const int* in_sizes, int n_in,
                              void* d_out, int out_size, void* d_ws, size_t ws_size,
                              hipStream_t stream) {
  (void)in_sizes; (void)n_in; (void)ws_size;
  const float* x        = (const float*)d_in[0];
  const int*   ei       = (const int*)d_in[1];
  const int*   src      = ei;
  const int*   dst      = ei + N_EDGES;
  const int*   batch    = (const int*)d_in[2];
  const float* enc_W1   = (const float*)d_in[3];
  const float* enc_b1   = (const float*)d_in[4];
  const float* enc_W2   = (const float*)d_in[5];
  const float* enc_b2   = (const float*)d_in[6];
  const float* rt_W1    = (const float*)d_in[7];
  const float* rt_b1    = (const float*)d_in[8];
  const float* rt_W2    = (const float*)d_in[9];
  const float* rt_b2    = (const float*)d_in[10];
  const float* centers  = (const float*)d_in[11];
  const float* s_relW   = (const float*)d_in[12];
  const float* s_relB   = (const float*)d_in[13];
  const float* s_rootW  = (const float*)d_in[14];
  const float* m_relW   = (const float*)d_in[15];
  const float* m_relB   = (const float*)d_in[16];
  const float* m_rootW  = (const float*)d_in[17];
  const float* e_relW   = (const float*)d_in[18];
  const float* e_relB   = (const float*)d_in[19];
  const float* e_rootW  = (const float*)d_in[20];

  char* ws = (char*)d_ws;
  const size_t NB = (size_t)N_NODES * HID * sizeof(float);  // 51.2 MB
  float* h     = (float*)(ws + 0 * NB);
  float* aggH  = (float*)(ws + 1 * NB);
  float* z1    = (float*)(ws + 2 * NB);
  float* z2    = (float*)(ws + 3 * NB);
  float* aggT  = (float*)(ws + 4 * NB);
  float* wgt   = (float*)(ws + 5 * NB);                    // N*4 f32
  char*  stats = ws + 5 * NB + (size_t)N_NODES * 4 * sizeof(float);
  float* ncnt  = (float*)(stats + 0);
  float* ecnt  = (float*)(stats + 256);
  float* gfn   = (float*)(stats + 512);
  float* lnn   = (float*)(stats + 1280);
  _Float16* packed = (_Float16*)(stats + 4096);            // 16 * 16384 halfs

  float* pred = (float*)d_out;

  // init
  hipMemsetAsync(stats, 0, 512, stream);
  hipMemsetAsync(pred, 0, (size_t)out_size * sizeof(float), stream);

  // graph stats
  count_nodes_kernel<<<(N_NODES + 255) / 256, 256, 0, stream>>>(batch, ncnt);
  count_edges_kernel<<<(N_EDGES + 255) / 256, 256, 0, stream>>>(src, batch, ecnt);
  graph_stats_kernel<<<1, 64, 0, stream>>>(ncnt, ecnt, gfn, lnn);

  // encoder + router
  encoder_kernel<<<N_NODES, 128, 0, stream>>>(x, enc_W1, enc_b1, enc_W2, enc_b2, h);
  router_kernel<<<N_NODES, 128, 0, stream>>>(h, batch, gfn, lnn, rt_W1, rt_b1, rt_W2, rt_b2,
                                             centers, wgt);

  // pack expert weights to WMMA B-fragment layout (f16)
  pack_weights_kernel<<<(16 * 16384) / 256, 256, 0, stream>>>(s_relW, s_rootW, m_relW, m_rootW,
                                                              packed);

  // shared start-layer aggregation (identical for all experts)
  hipMemsetAsync(aggH, 0, NB, stream);
  agg_scatter_kernel<<<(unsigned)((size_t)N_EDGES * 32 / 256), 256, 0, stream>>>(h, aggH, src, dst);

  const int gemmBlocks = (N_NODES / 16 + 7) / 8;  // 782 blocks, 8 row tiles each
  for (int e = 0; e < 4; ++e) {
    // start: z1 = relu(aggH @ s_relW[e] + h @ s_rootW[e] + s_relB[e])
    gemm_dual_kernel<<<gemmBlocks, 256, 0, stream>>>(
        aggH, packed + (size_t)(e * 4 + 0) * 16384,
        h,    packed + (size_t)(e * 4 + 1) * 16384,
        s_relB + (size_t)e * HID, z1, 1);

    // mid: aggT = segsum(z1); z2 = relu(aggT @ m_relW[e] + z1 @ m_rootW[e] + m_relB[e])
    hipMemsetAsync(aggT, 0, NB, stream);
    agg_scatter_kernel<<<(unsigned)((size_t)N_EDGES * 32 / 256), 256, 0, stream>>>(z1, aggT, src, dst);
    gemm_dual_kernel<<<gemmBlocks, 256, 0, stream>>>(
        aggT, packed + (size_t)(e * 4 + 2) * 16384,
        z1,   packed + (size_t)(e * 4 + 3) * 16384,
        m_relB + (size_t)e * HID, z2, 1);

    // end: aggT = segsum(z2); pred += w_e * (aggT @ e_relW[e] + e_relB[e] + z2 @ e_rootW[e])
    hipMemsetAsync(aggT, 0, NB, stream);
    agg_scatter_kernel<<<(unsigned)((size_t)N_EDGES * 32 / 256), 256, 0, stream>>>(z2, aggT, src, dst);
    end_layer_kernel<<<(N_NODES + 255) / 256, 256, 0, stream>>>(
        aggT, z2, e_relW + (size_t)e * HID * 4, e_relB + (size_t)e * 4,
        e_rootW + (size_t)e * HID * 4, wgt, e, pred);
  }
}
